// LinformerAttentionBlock2d_7988639171307
// MI455X (gfx1250) — matrix operand order repaired and checked by reading, not compiled, not run on hardware
//
#include <hip/hip_runtime.h>

// ---------------------------------------------------------------------------
// Linformer attention block for MI455X (gfx1250, wave32, WMMA).
// All matmuls run on v_wmma_f32_16x16x32_bf16 (fp32 accumulate).
// Assumed WMMA VGPR layouts (CDNA5 ISA §7.12.2), lane l, lm=l&15,
// lh=l>>4:
//   A (16x32 bf16): row m=lm; element i: K = (i<8 ? lh*8+i : 16+lh*8+(i-8))
//   B (32x16 bf16): col n=lm; element i: K = (i<8 ? lh*8+i : 16+lh*8+(i-8))
//   C/D (16x16 f32): col n=lm; vgpr r: row m = r + lh*8
// ---------------------------------------------------------------------------

typedef __bf16 bf16;
typedef __attribute__((ext_vector_type(16))) __bf16 v16bf;
typedef __attribute__((ext_vector_type(8)))  __bf16 v8bf;
typedef __attribute__((ext_vector_type(8)))  float  v8f;

__device__ __forceinline__ v8f wmma_bf16(v16bf a, v16bf b, v8f c) {
  return __builtin_amdgcn_wmma_f32_16x16x32_bf16(false, a, false, b,
                                                 (short)0, c, false, false);
}

__device__ __forceinline__ v16bf frag2(const bf16* p0, const bf16* p1) {
  v8bf x = *(const v8bf*)p0;
  v8bf y = *(const v8bf*)p1;
  return __builtin_shufflevector(x, y, 0, 1, 2, 3, 4, 5, 6, 7,
                                 8, 9, 10, 11, 12, 13, 14, 15);
}

// ---------------------------------------------------------------------------
// Kernel 1: GroupNorm. One block per (batch, group): 64 ch x 1024 px = 65536.
// Writes xn fp32 (residual path) and xn bf16 (GEMM path), layout [b][c][n].
// ---------------------------------------------------------------------------
__global__ __launch_bounds__(256) void groupnorm_kernel(
    const float* __restrict__ x, const float* __restrict__ gamma,
    const float* __restrict__ beta, float* __restrict__ xn_f,
    bf16* __restrict__ xn_b) {
  const int bg = blockIdx.x;           // b*8 + g
  const int g = bg & 7;
  const int tid = threadIdx.x;
  const long base = (long)bg * 65536;  // groups are contiguous 64ch x 1024px

  float s = 0.f, ss = 0.f;
  for (int i = tid; i < 65536; i += 256) {
    float v = x[base + i];
    s += v;
    ss += v * v;
  }
  __shared__ float r1[256], r2[256];
  r1[tid] = s;
  r2[tid] = ss;
  __syncthreads();
  for (int st = 128; st > 0; st >>= 1) {
    if (tid < st) {
      r1[tid] += r1[tid + st];
      r2[tid] += r2[tid + st];
    }
    __syncthreads();
  }
  const float mu = r1[0] * (1.f / 65536.f);
  const float var = r2[0] * (1.f / 65536.f) - mu * mu;
  const float rstd = rsqrtf(var + 1e-5f);

  for (int i = tid; i < 65536; i += 256) {
    int c = g * 64 + (i >> 10);
    float v = (x[base + i] - mu) * rstd * gamma[c] + beta[c];
    xn_f[base + i] = v;
    xn_b[base + i] = (bf16)v;
  }
}

// ---------------------------------------------------------------------------
// Kernel 2: fp32 -> bf16 conversion (weights, E, F).
// ---------------------------------------------------------------------------
__global__ void f32_to_bf16_kernel(const float* __restrict__ in,
                                   bf16* __restrict__ out, int n) {
  int i = blockIdx.x * blockDim.x + threadIdx.x;
  if (i < n) out[i] = (bf16)in[i];
}

// ---------------------------------------------------------------------------
// Kernel 3: generic bf16 WMMA GEMM, out[m][n] = sum_k A[m][k]*B[k][n].
// Block tile 64x64, 256 threads = 8 waves, each wave two 16x16 accumulators.
// BT=true: B stored as [n][k] row-major.  BIAS: 0 none, 1 per-row, 2 per-col.
// ---------------------------------------------------------------------------
template <int BIAS, bool BT, bool OF32, bool RES>
__global__ __launch_bounds__(256) void gemm_bf16_kernel(
    const bf16* __restrict__ A, int lda, long sA,
    const bf16* __restrict__ Bm, int ldb, long sB,
    const float* __restrict__ bias,
    bf16* __restrict__ oB, float* __restrict__ oF, int ldo, long sO,
    const float* __restrict__ resid, long sR, int Kd) {
  const int m0 = blockIdx.x * 64;
  const int n0 = blockIdx.y * 64;
  const int bz = blockIdx.z;
  const int tid = threadIdx.x;
  const int w = tid >> 5, l = tid & 31, lm = l & 15, lh = l >> 4;
  const int tm = w & 3, tn0 = (w >> 2) * 2;

  __shared__ __align__(16) bf16 Al[64][40];  // [m][k], padded stride
  __shared__ __align__(16) bf16 Bl[64][40];  // [n][k], padded stride

  const bf16* Ab = A + (long)bz * sA;
  const bf16* Bb = Bm + (long)bz * sB;

  v8f acc0 = {0, 0, 0, 0, 0, 0, 0, 0};
  v8f acc1 = {0, 0, 0, 0, 0, 0, 0, 0};

  for (int k0 = 0; k0 < Kd; k0 += 32) {
    __syncthreads();
    for (int e = tid; e < 64 * 32; e += 256) {
      int r = e >> 5, c = e & 31;  // coalesced along k
      Al[r][c] = Ab[(long)(m0 + r) * lda + k0 + c];
    }
    if (BT) {
      for (int e = tid; e < 64 * 32; e += 256) {
        int r = e >> 5, c = e & 31;  // coalesced along k
        Bl[r][c] = Bb[(long)(n0 + r) * ldb + k0 + c];
      }
    } else {
      for (int e = tid; e < 64 * 32; e += 256) {
        int r = e & 63, c = e >> 6;  // coalesced along n, transpose into LDS
        Bl[r][c] = Bb[(long)(k0 + c) * ldb + n0 + r];
      }
    }
    __syncthreads();

    const bf16* ar = &Al[tm * 16 + lm][0];
    v16bf af = frag2(ar + lh * 8, ar + 16 + lh * 8);
    const bf16* b0r = &Bl[tn0 * 16 + lm][0];
    const bf16* b1r = &Bl[(tn0 + 1) * 16 + lm][0];
    v16bf bf0 = frag2(b0r + lh * 8, b0r + 16 + lh * 8);
    v16bf bf1 = frag2(b1r + lh * 8, b1r + 16 + lh * 8);
    acc0 = wmma_bf16(af, bf0, acc0);
    acc1 = wmma_bf16(af, bf1, acc1);
  }

  const long ob = (long)bz * sO;
#pragma unroll
  for (int r = 0; r < 8; ++r) {
    int row = m0 + tm * 16 + r + lh * 8;
    int c0 = n0 + tn0 * 16 + lm;
    int c1 = c0 + 16;
    float v0 = acc0[r], v1 = acc1[r];
    if (BIAS == 1) {
      float bb = bias[row];
      v0 += bb;
      v1 += bb;
    }
    if (BIAS == 2) {
      v0 += bias[c0];
      v1 += bias[c1];
    }
    if (RES) {
      const float* rb = resid + (long)bz * sR;
      v0 += rb[(long)row * ldo + c0];
      v1 += rb[(long)row * ldo + c1];
    }
    if (OF32) {
      oF[ob + (long)row * ldo + c0] = v0;
      oF[ob + (long)row * ldo + c1] = v1;
    } else {
      oB[ob + (long)row * ldo + c0] = (bf16)v0;
      oB[ob + (long)row * ldo + c1] = (bf16)v1;
    }
  }
}

// ---------------------------------------------------------------------------
// Kernel 4: fused attention: scores = qK'/8, softmax over K=256, out = A*vp.
// Block = 128 threads (4 waves) handles 32 query rows for one (b, head).
// qkv: [b][3C][n]; kp/vp: [b][c][K] (c = h*64+d); out: [b][n][c] bf16.
// Static LDS = 64,512 B (kept under 64KB).
// ---------------------------------------------------------------------------
__global__ __launch_bounds__(128) void attn_fused_kernel(
    const bf16* __restrict__ qkv, const bf16* __restrict__ kp,
    const bf16* __restrict__ vp, bf16* __restrict__ aout) {
  const int n0 = blockIdx.x * 32;
  const int h = blockIdx.y;
  const int b = blockIdx.z;
  const int tid = threadIdx.x;
  const int w = tid >> 5, l = tid & 31, lm = l & 15, lh = l >> 4;
  const int rh = w >> 1;        // row-tile (0/1) owned by this wave
  const int halfsel = w & 1;    // column-half owned by this wave

  __shared__ __align__(16) bf16 q_lds[32][72];    // [n][d]
  __shared__ __align__(16) float sc[32][256];     // fp32 scores
  __shared__ __align__(16) bf16 attn_l[32][256];  // bf16 softmax(A)
  __shared__ __align__(16) bf16 kc_lds[32][72];   // kp chunk [kk][d]
  __shared__ __align__(16) bf16 vc_lds[64][40];   // vp chunk [d][kk]
  __shared__ float redm[32][4];
  __shared__ float reds[32][4];

  const long qbase = ((long)b * 3 * 512 + h * 64) * 1024;
  const long kvbase = ((long)b * 512 + h * 64) * 256;

  // stage q tile: q_lds[n][d] = qkv[b][h*64+d][n0+n]
  for (int e = tid; e < 32 * 64; e += 128) {
    int nn = e & 31, d = e >> 5;  // coalesced along n
    q_lds[nn][d] = qkv[qbase + (long)d * 1024 + n0 + nn];
  }

  // ---- phase 1: scores = (q @ kp^T) / sqrt(DH), DH=64 contraction ----
  for (int kcb = 0; kcb < 256; kcb += 32) {
    __syncthreads();
    for (int e = tid; e < 32 * 64; e += 128) {
      int kk = e & 31, d = e >> 5;  // coalesced along kk
      kc_lds[kk][d] = kp[kvbase + (long)d * 256 + kcb + kk];
    }
    __syncthreads();
    const bf16* qr = &q_lds[rh * 16 + lm][0];
    v16bf a0 = frag2(qr + lh * 8, qr + 16 + lh * 8);        // d 0..31
    v16bf a1 = frag2(qr + 32 + lh * 8, qr + 48 + lh * 8);   // d 32..63
    const bf16* kr = &kc_lds[halfsel * 16 + lm][0];
    v16bf b0 = frag2(kr + lh * 8, kr + 16 + lh * 8);
    v16bf b1 = frag2(kr + 32 + lh * 8, kr + 48 + lh * 8);
    v8f acc = {0, 0, 0, 0, 0, 0, 0, 0};
    acc = wmma_bf16(a0, b0, acc);
    acc = wmma_bf16(a1, b1, acc);
#pragma unroll
    for (int r = 0; r < 8; ++r)
      sc[rh * 16 + r + lh * 8][kcb + halfsel * 16 + lm] = acc[r] * 0.125f;
  }
  __syncthreads();

  // ---- phase 2: softmax over 256, 4 threads per row ----
  {
    int r = tid >> 2, j = tid & 3;
    float* row = &sc[r][j * 64];
    float mx = -3.0e38f;
    for (int i = 0; i < 64; ++i) mx = fmaxf(mx, row[i]);
    redm[r][j] = mx;
    __syncthreads();
    mx = fmaxf(fmaxf(redm[r][0], redm[r][1]), fmaxf(redm[r][2], redm[r][3]));
    float s = 0.f;
    for (int i = 0; i < 64; ++i) {
      float p = __expf(row[i] - mx);
      row[i] = p;
      s += p;
    }
    reds[r][j] = s;
    __syncthreads();
    float inv = 1.f / (reds[r][0] + reds[r][1] + reds[r][2] + reds[r][3]);
    bf16* arow = &attn_l[r][j * 64];
    for (int i = 0; i < 64; ++i) arow[i] = (bf16)(row[i] * inv);
  }

  // ---- phase 3: out[n][d] = sum_kk attn[n][kk] * vp[kk][d] ----
  const int d0 = halfsel * 32;
  v8f o0 = {0, 0, 0, 0, 0, 0, 0, 0};
  v8f o1 = {0, 0, 0, 0, 0, 0, 0, 0};
  for (int kcb = 0; kcb < 256; kcb += 32) {
    __syncthreads();
    for (int e = tid; e < 64 * 32; e += 128) {
      int kk = e & 31, d = e >> 5;  // coalesced along kk
      vc_lds[d][kk] = vp[kvbase + (long)d * 256 + kcb + kk];
    }
    __syncthreads();
    const bf16* ar = &attn_l[rh * 16 + lm][kcb];
    v16bf af = frag2(ar + lh * 8, ar + 16 + lh * 8);
    const bf16* v0r = &vc_lds[d0 + lm][0];
    const bf16* v1r = &vc_lds[d0 + 16 + lm][0];
    v16bf b0 = frag2(v0r + lh * 8, v0r + 16 + lh * 8);
    v16bf b1 = frag2(v1r + lh * 8, v1r + 16 + lh * 8);
    o0 = wmma_bf16(af, b0, o0);
    o1 = wmma_bf16(af, b1, o1);
  }

  const long obase = ((long)b * 1024 + n0) * 512 + h * 64;
#pragma unroll
  for (int r = 0; r < 8; ++r) {
    int nn = rh * 16 + r + lh * 8;
    aout[obase + (long)nn * 512 + d0 + lm] = (bf16)o0[r];
    aout[obase + (long)nn * 512 + d0 + 16 + lm] = (bf16)o1[r];
  }
}

// ---------------------------------------------------------------------------
// Host launcher.
// ---------------------------------------------------------------------------
extern "C" void kernel_launch(void* const* d_in, const int* in_sizes, int n_in,
                              void* d_out, int out_size, void* d_ws,
                              size_t ws_size, hipStream_t stream) {
  (void)in_sizes; (void)n_in; (void)out_size; (void)ws_size;

  const float* x      = (const float*)d_in[0];
  const float* gamma  = (const float*)d_in[1];
  const float* beta   = (const float*)d_in[2];
  const float* w_qkv  = (const float*)d_in[3];
  const float* b_qkv  = (const float*)d_in[4];
  const float* E      = (const float*)d_in[5];
  const float* F      = (const float*)d_in[6];
  const float* w_o    = (const float*)d_in[7];
  const float* b_o    = (const float*)d_in[8];
  const float* w_proj = (const float*)d_in[9];
  const float* b_proj = (const float*)d_in[10];
  float* out = (float*)d_out;

  const int Bn = 16, C = 512, N = 1024, K = 256;

  char* ws = (char*)d_ws;
  size_t off = 0;
  auto alloc = [&](size_t bytes) -> void* {
    void* p = ws + off;
    off = (off + bytes + 255) & ~(size_t)255;
    return p;
  };

  float* xn_f  = (float*)alloc(sizeof(float) * (size_t)Bn * C * N);  // 33.5MB
  bf16* xn_b   = (bf16*)alloc(2ull * Bn * C * N);
  bf16* qkv_b  = (bf16*)alloc(2ull * Bn * 3 * C * N);
  bf16* kp_b   = (bf16*)alloc(2ull * Bn * C * K);
  bf16* vp_b   = (bf16*)alloc(2ull * Bn * C * K);
  bf16* ao_b   = (bf16*)alloc(2ull * Bn * N * C);
  bf16* o2_b   = (bf16*)alloc(2ull * Bn * N * C);
  bf16* wqkv_b = (bf16*)alloc(2ull * 3 * C * C);
  bf16* wo_b   = (bf16*)alloc(2ull * C * C);
  bf16* wp_b   = (bf16*)alloc(2ull * C * C);
  bf16* E_b    = (bf16*)alloc(2ull * N * K);
  bf16* F_b    = (bf16*)alloc(2ull * N * K);

  // 1) GroupNorm
  groupnorm_kernel<<<dim3(Bn * 8), dim3(256), 0, stream>>>(x, gamma, beta,
                                                           xn_f, xn_b);

  // 2) weight / projection conversions to bf16
  auto conv = [&](const float* s, bf16* d, int n) {
    f32_to_bf16_kernel<<<dim3((n + 255) / 256), dim3(256), 0, stream>>>(s, d,
                                                                        n);
  };
  conv(w_qkv, wqkv_b, 3 * C * C);
  conv(w_o, wo_b, C * C);
  conv(w_proj, wp_b, C * C);
  conv(E, E_b, N * K);
  conv(F, F_b, N * K);

  // 3) qkv[b][o][n] = w_qkv @ xn + b_qkv   (M=1536, N=1024, K=512)
  gemm_bf16_kernel<1, false, false, false>
      <<<dim3(24, 16, Bn), dim3(256), 0, stream>>>(
          wqkv_b, C, 0, xn_b, N, (long)C * N, b_qkv, qkv_b, nullptr, N,
          (long)3 * C * N, nullptr, 0, C);

  // 4) kp^T[b][c][k] = k[c][n] @ E[n][k]   (M=512, N=256, K=1024)
  gemm_bf16_kernel<0, false, false, false>
      <<<dim3(8, 4, Bn), dim3(256), 0, stream>>>(
          qkv_b + (size_t)C * N, N, (long)3 * C * N, E_b, K, 0, nullptr, kp_b,
          nullptr, K, (long)C * K, nullptr, 0, N);

  // 5) vp^T[b][c][k] = v[c][n] @ F[n][k]
  gemm_bf16_kernel<0, false, false, false>
      <<<dim3(8, 4, Bn), dim3(256), 0, stream>>>(
          qkv_b + 2ull * C * N, N, (long)3 * C * N, F_b, K, 0, nullptr, vp_b,
          nullptr, K, (long)C * K, nullptr, 0, N);

  // 6) fused attention -> ao_b [b][n][c]
  attn_fused_kernel<<<dim3(32, 8, Bn), dim3(128), 0, stream>>>(qkv_b, kp_b,
                                                               vp_b, ao_b);

  // 7) o2[b][n][o] = ao @ w_o^T + b_o   (M=1024, N=512, K=512; per-col bias)
  gemm_bf16_kernel<2, true, false, false>
      <<<dim3(16, 8, Bn), dim3(256), 0, stream>>>(
          ao_b, C, (long)N * C, wo_b, C, 0, b_o, o2_b, nullptr, C,
          (long)N * C, nullptr, 0, C);

  // 8) out[b][c][n] = w_proj @ o2^T + b_proj + xn   (M=512, N=1024, K=512)
  gemm_bf16_kernel<1, true, true, true>
      <<<dim3(8, 16, Bn), dim3(256), 0, stream>>>(
          wp_b, C, 0, o2_b, C, (long)N * C, b_proj, nullptr, out, N,
          (long)C * N, xn_f, (long)C * N, C);
}
